// VectorQuantizerEMA_83648783057671
// MI455X (gfx1250) — matrix operand order repaired and checked by reading, not compiled, not run on hardware
//
#include <hip/hip_runtime.h>
#include <stdint.h>

// ---------------------------------------------------------------------------
// Problem constants (from reference): z [8,8,512,256] f32, codebook [8192,256]
// ---------------------------------------------------------------------------
#define NROWS   32768      // 8*8*512
#define DDIM    256
#define KCODES  8192
#define NELEM   (NROWS * DDIM)   // 8388608
#define TCODES  32               // codes staged per LDS tile
#define NTILES  (KCODES / TCODES)

typedef __attribute__((ext_vector_type(16))) __bf16 v16bf;
typedef __attribute__((ext_vector_type(8)))  float  v8f;

union Frag32B { uint4 u[2]; v16bf v; };

__device__ __forceinline__ unsigned short f2bf(float f) {
    unsigned int u = __float_as_uint(f);
    u += 0x7FFFu + ((u >> 16) & 1u);          // round-to-nearest-even
    return (unsigned short)(u >> 16);
}

__device__ __forceinline__ uint4 pack8(const float* f) {
    uint4 r;
    r.x = (unsigned)f2bf(f[0]) | ((unsigned)f2bf(f[1]) << 16);
    r.y = (unsigned)f2bf(f[2]) | ((unsigned)f2bf(f[3]) << 16);
    r.z = (unsigned)f2bf(f[4]) | ((unsigned)f2bf(f[5]) << 16);
    r.w = (unsigned)f2bf(f[6]) | ((unsigned)f2bf(f[7]) << 16);
    return r;
}

// ---------------------------------------------------------------------------
// CDNA5 async global->LDS copy (ASYNCcnt-tracked; bypasses VGPRs).
// Per CDNA5 ISA: INST_OFFSET is added to BOTH the LDS address
// and the global address, so one immediate advances both sides.
// ---------------------------------------------------------------------------
template <int OFS>
__device__ __forceinline__ void async_copy_b128(unsigned lds_addr, const void* gaddr) {
    asm volatile("global_load_async_to_lds_b128 %0, %1, off offset:%2"
                 :: "v"(lds_addr), "v"(gaddr), "i"(OFS) : "memory");
}
__device__ __forceinline__ void wait_asynccnt0() {
    asm volatile("s_wait_asynccnt 0x0" ::: "memory");
}

// ---------------------------------------------------------------------------
// Kernel: zero the loss accumulator (ws is poisoned, must init every call)
// ---------------------------------------------------------------------------
__global__ void k_init(float* lacc) { if (threadIdx.x == 0) lacc[0] = 0.0f; }

// ---------------------------------------------------------------------------
// Kernel: convert z (f32 -> bf16), 8 elems/thread
// ---------------------------------------------------------------------------
__global__ void k_prep_z(const float* __restrict__ z, unsigned short* __restrict__ zb) {
    int t = blockIdx.x * blockDim.x + threadIdx.x;     // 0 .. NELEM/8-1
    float f[8];
    const float4* zf = (const float4*)z + (size_t)t * 2;
    *(float4*)(f)     = zf[0];
    *(float4*)(f + 4) = zf[1];
    ((uint4*)zb)[t] = pack8(f);
}

// ---------------------------------------------------------------------------
// Kernel: convert codebook (f32 -> bf16) + per-code squared norms.
// One wave per codebook row (8 rows / 256-thread block).
// ---------------------------------------------------------------------------
__global__ void k_prep_cb(const float* __restrict__ cb,
                          unsigned short* __restrict__ cbb,
                          float* __restrict__ cbnorm) {
    int wave = threadIdx.x >> 5, lane = threadIdx.x & 31;
    int row  = blockIdx.x * 8 + wave;
    float f[8];
    const float4* src = (const float4*)(cb + (size_t)row * DDIM) + lane * 2;
    *(float4*)(f)     = src[0];
    *(float4*)(f + 4) = src[1];
    float ssq = 0.f;
#pragma unroll
    for (int i = 0; i < 8; ++i) ssq += f[i] * f[i];
    ((uint4*)(cbb + (size_t)row * DDIM))[lane] = pack8(f);
#pragma unroll
    for (int off = 16; off >= 1; off >>= 1) ssq += __shfl_xor(ssq, off, 32);
    if (lane == 0) cbnorm[row] = ssq;
}

// ---------------------------------------------------------------------------
// Main fused kernel: S = z_bf16 . cb_bf16^T via v_wmma_f32_16x16x32_bf16,
// fused running argmin of (||cb||^2 - 2*S) per row.
// 256 blocks x 256 threads (8 waves); wave w owns rows [blk*128 + w*16, +16).
// 32-code x 256-d codebook tiles double-buffered in LDS via async copies:
// staging of tile t+1 overlaps the 16 WMMAs of tile t.
// ---------------------------------------------------------------------------
__global__ void __launch_bounds__(256)
k_argmin(const unsigned short* __restrict__ zb,
         const unsigned short* __restrict__ cbb,
         const float* __restrict__ cbnorm,
         int* __restrict__ codes) {
    __shared__ uint4 sB[2][TCODES * DDIM / 8];   // 2 x 16 KB
    const int tid   = threadIdx.x;
    const int wave  = tid >> 5;
    const int lane  = tid & 31;
    const int hsel  = lane >> 4;                 // half-wave select
    const int l15   = lane & 15;
    const int rbase = blockIdx.x * 128 + wave * 16;

    // A fragments: 16 rows x 256 d, ISA 16-bit A layout:
    // lanes 0-15: K = [0..7] and [16..23]; lanes 16-31: K = [8..15] and [24..31]
    Frag32B aF[8];
    {
        const unsigned short* rowp = zb + (size_t)(rbase + l15) * DDIM;
#pragma unroll
        for (int c = 0; c < 8; ++c) {
            int d0 = c * 32 + hsel * 8;
            aF[c].u[0] = *(const uint4*)(rowp + d0);
            aF[c].u[1] = *(const uint4*)(rowp + d0 + 16);
        }
    }

    float minv[8];
    int   mini[8];
#pragma unroll
    for (int i = 0; i < 8; ++i) { minv[i] = 3.4e38f; mini[i] = 0; }

    const uint4* cbq = (const uint4*)cbb;   // 32 uint4 per code row, 1024/tile

    // Per-thread staging slice: 64 contiguous bytes (4 x b128 async copies)
    unsigned ldsA0 = (unsigned)(uintptr_t)(&sB[0][tid * 4]);
    unsigned ldsA1 = (unsigned)(uintptr_t)(&sB[1][tid * 4]);

    // Prologue: stage tile 0 into buffer 0
    {
        const void* g = cbq + (size_t)tid * 4;
        async_copy_b128<0>(ldsA0, g);
        async_copy_b128<16>(ldsA0, g);
        async_copy_b128<32>(ldsA0, g);
        async_copy_b128<48>(ldsA0, g);
    }
    wait_asynccnt0();
    __syncthreads();

    int buf = 0;
    for (int t = 0; t < NTILES; ++t) {
        // Kick off async staging of tile t+1 into the other buffer
        if (t + 1 < NTILES) {
            const void* g = cbq + (size_t)(t + 1) * (TCODES * DDIM / 8) + tid * 4;
            unsigned dst = buf ? ldsA0 : ldsA1;
            async_copy_b128<0>(dst, g);
            async_copy_b128<16>(dst, g);
            async_copy_b128<32>(dst, g);
            async_copy_b128<48>(dst, g);
        }

        const unsigned short* sb = (const unsigned short*)sB[buf];
#pragma unroll
        for (int s = 0; s < 2; ++s) {            // two 16-code sub-tiles
            // B fragments: column = code (lane&15);
            // lanes 0-15 hold K [0..7],[16..23], lanes 16-31 hold K [8..15],[24..31] of each chunk
            const unsigned short* colp = sb + (s * 16 + l15) * DDIM + hsel * 8;
            Frag32B bF[8];
#pragma unroll
            for (int c = 0; c < 8; ++c) {
                bF[c].u[0] = *(const uint4*)(colp + c * 32);
                bF[c].u[1] = *(const uint4*)(colp + c * 32 + 16);
            }
            float cbn = cbnorm[t * TCODES + s * 16 + l15];

            v8f acc = {};
#pragma unroll
            for (int c = 0; c < 8; ++c)
                acc = __builtin_amdgcn_wmma_f32_16x16x32_bf16(
                    false, aF[c].v, false, bF[c].v, (short)0, acc, false, false);

            // dist proxy = ||cb||^2 - 2*S ; strict < keeps lowest code id
            const int codeid = t * TCODES + s * 16 + l15;
#pragma unroll
            for (int i = 0; i < 8; ++i) {
                float dist = fmaf(-2.0f, acc[i], cbn);
                bool take  = dist < minv[i];
                minv[i] = take ? dist   : minv[i];
                mini[i] = take ? codeid : mini[i];
            }
        }

        wait_asynccnt0();       // this wave's staging of t+1 has landed
        __syncthreads();        // everyone done reading buf & staging buf^1
        buf ^= 1;
    }

    // Cross-lane argmin within each 16-lane half (C layout: lanes 0-15 rows 0-7,
    // lanes 16-31 rows 8-15). Tie-break: lower code index, matching jnp.argmin.
#pragma unroll
    for (int off = 8; off >= 1; off >>= 1) {
#pragma unroll
        for (int i = 0; i < 8; ++i) {
            float ov = __shfl_xor(minv[i], off, 32);
            int   oi = __shfl_xor(mini[i], off, 32);
            bool take = (ov < minv[i]) || ((ov == minv[i]) && (oi < mini[i]));
            minv[i] = take ? ov : minv[i];
            mini[i] = take ? oi : mini[i];
        }
    }
    if (l15 == 0) {
        int base = rbase + hsel * 8;
#pragma unroll
        for (int i = 0; i < 8; ++i) codes[base + i] = mini[i];
    }
}

// ---------------------------------------------------------------------------
// Gather: z_q = codebook[code] (f32), accumulate sum((z_q - z)^2), emit codes.
// One wave per row; 8 rows per 256-thread block.
// ---------------------------------------------------------------------------
__global__ void k_gather(const float* __restrict__ cb,
                         const float* __restrict__ z,
                         const int* __restrict__ codes,
                         float* __restrict__ zq,
                         int* __restrict__ codes_out,
                         float* __restrict__ lacc) {
    int wave = threadIdx.x >> 5, lane = threadIdx.x & 31;
    int row  = blockIdx.x * 8 + wave;
    int code = codes[row];
    const float4* src = (const float4*)(cb + (size_t)code * DDIM) + lane * 2;
    const float4* zs  = (const float4*)(z  + (size_t)row  * DDIM) + lane * 2;
    float4*       dst = (float4*)(zq + (size_t)row * DDIM) + lane * 2;
    float s = 0.f;
#pragma unroll
    for (int j = 0; j < 2; ++j) {
        float4 a = src[j], b = zs[j];
        dst[j] = a;
        float dx = a.x - b.x, dy = a.y - b.y, dz = a.z - b.z, dw = a.w - b.w;
        s += dx * dx + dy * dy + dz * dz + dw * dw;
    }
#pragma unroll
    for (int off = 16; off >= 1; off >>= 1) s += __shfl_xor(s, off, 32);
    if (lane == 0) {
        atomicAdd(lacc, s);
        codes_out[row] = code;
    }
}

__global__ void k_finalize(const float* __restrict__ lacc, float* __restrict__ loss_out) {
    if (threadIdx.x == 0)
        loss_out[0] = 0.5f * lacc[0] * (1.0f / (float)NELEM);
}

// ---------------------------------------------------------------------------
// Host-side launcher
// ---------------------------------------------------------------------------
extern "C" void kernel_launch(void* const* d_in, const int* in_sizes, int n_in,
                              void* d_out, int out_size, void* d_ws, size_t ws_size,
                              hipStream_t stream) {
    const float* z  = (const float*)d_in[0];   // [NROWS, DDIM]
    const float* cb = (const float*)d_in[1];   // [KCODES, DDIM]

    // Workspace layout (16B aligned)
    char* ws = (char*)d_ws;
    unsigned short* cb_bf16 = (unsigned short*)ws;                    ws += (size_t)KCODES * DDIM * 2;
    unsigned short* z_bf16  = (unsigned short*)ws;                    ws += (size_t)NROWS  * DDIM * 2;
    float* cbnorm           = (float*)ws;                             ws += (size_t)KCODES * 4;
    int*   codes_ws         = (int*)ws;                               ws += (size_t)NROWS  * 4;
    float* lacc             = (float*)ws;

    // Output layout: z_q_st [NELEM] f32 | total_loss [1] f32 | codes [NROWS] i32
    float* zq_out    = (float*)d_out;
    float* loss_out  = zq_out + NELEM;
    int*   codes_out = (int*)(zq_out + NELEM + 1);

    k_init    <<<1, 64, 0, stream>>>(lacc);
    k_prep_z  <<<NELEM / (256 * 8), 256, 0, stream>>>(z, z_bf16);
    k_prep_cb <<<KCODES / 8,        256, 0, stream>>>(cb, cb_bf16, cbnorm);
    k_argmin  <<<NROWS / 128,       256, 0, stream>>>(z_bf16, cb_bf16, cbnorm, codes_ws);
    k_gather  <<<NROWS / 8,         256, 0, stream>>>(cb, z, codes_ws, zq_out, codes_out, lacc);
    k_finalize<<<1, 64, 0, stream>>>(lacc, loss_out);
}
